// NeuralODE_63934883168374
// MI455X (gfx1250) — matrix operand order repaired and checked by reading, not compiled, not run on hardware
//
#include <hip/hip_runtime.h>
#include <hip/hip_bf16.h>
#include <math.h>

// ---------------------------------------------------------------------------
// Neural ODE (Heun, 10 steps) fully fused for MI455X (gfx1250).
//
// Reasoning from MI455X specs:
//  * 687 GFLOP total vs ~34 MB HBM traffic -> massively compute bound.
//  * Weights (8.4 MB f32, 4 MB as f16) sit in the 192 MB L2 for the whole
//    kernel; every workgroup streams them from L2 each f-eval.
//  * Batch rows are independent ODEs -> one kernel, no global sync: each
//    workgroup integrates 32 rows for all 10 steps with y held in VGPRs.
//  * Matrix math on the wave32 WMMA path: v_wmma_f32_16x16x32_f16
//    (f16 A/B, f32 accumulate). All reductions/updates stay f32.
// ---------------------------------------------------------------------------

#define BATCH   8192
#define DDIM    512
#define HDIM    2048
#define MT      32          // rows per workgroup
#define HC      64          // hidden-chunk per fused GEMM1->tanh->GEMM2 pass
#define NSTEPS  10
#define DT      0.1f

typedef __attribute__((ext_vector_type(16))) _Float16 v16h;
typedef __attribute__((ext_vector_type(8)))  _Float16 v8h;
typedef __attribute__((ext_vector_type(8)))  float    v8f;

// Build a 16-half WMMA fragment from two 16-byte chunks (A-layout needs the
// two K-subchunks 16 halves apart; B-layout takes the same two subchunks).
__device__ __forceinline__ v16h ld_frag(const _Float16* p0, const _Float16* p1) {
    v8h lo = *(const v8h*)p0;
    v8h hi = *(const v8h*)p1;
    v16h r;
#pragma unroll
    for (int i = 0; i < 8; ++i) { r[i] = lo[i]; r[i + 8] = hi[i]; }
    return r;
}

// ---- prep: convert + transpose weights to f16, B-friendly layouts ----------
// W1  [D,H] f32 row-major  ->  W1T [H,D] f16 row-major  (GEMM1 B: N-major)
__global__ void cvt_w1t(const float* __restrict__ w1, _Float16* __restrict__ w1t) {
    int o = blockIdx.x * blockDim.x + threadIdx.x;     // o = h*512 + d
    if (o >= HDIM * DDIM) return;
    int h = o >> 9, d = o & 511;
    w1t[o] = (_Float16)w1[((long)d << 11) + h];
}
// W2  [H,D] f32 row-major  ->  W2T [D,H] f16 row-major  (GEMM2 B: N-major)
__global__ void cvt_w2t(const float* __restrict__ w2, _Float16* __restrict__ w2t) {
    int o = blockIdx.x * blockDim.x + threadIdx.x;     // o = d*2048 + h
    if (o >= DDIM * HDIM) return;
    int d = o >> 11, h = o & 2047;
    w2t[o] = (_Float16)w2[((long)h << 9) + d];
}

// ---- main fused integrator -------------------------------------------------
__global__ __launch_bounds__(256, 1)
void node_heun(const float* __restrict__ y0,
               const float* __restrict__ b1,
               const float* __restrict__ b2,
               const _Float16* __restrict__ w1t,   // [H, D]
               const _Float16* __restrict__ w2t,   // [D, H]
               float* __restrict__ yout)
{
    __shared__ __align__(16) _Float16 yh[MT * DDIM];   // 32 KB: current state, f16
    __shared__ __align__(16) _Float16 hb[MT * HC];     //  4 KB: tanh chunk, f16

    const int tid   = threadIdx.x;
    const int w     = tid >> 5;        // wave 0..7
    const int lane  = tid & 31;
    const int lhalf = lane >> 4;       // K-half select (wave32 WMMA layout)
    const int l16   = lane & 15;

    const int rt    = w >> 2;                 // row tile 0/1 (16 rows each)
    const int ct    = w & 3;                  // GEMM1 col quarter / k col block
    const int rowA  = rt * 16 + l16;          // A-fragment row (GEMM1)
    const int crow0 = rt * 16 + 8 * lhalf;    // C/D row base: vgpr i -> crow0+i
    const int kcol0 = ct * 128 + l16;         // k/y column base; tile t adds 16t

    const long grow0 = (long)blockIdx.x * MT;

    v8f  y[8];        // state, f32, C/D layout
    v8f  acc[8];      // f-eval accumulator (raw h@W2, pre-bias)
    float b2c[8];     // b2 per owned column (column is per-lane constant!)

#pragma unroll
    for (int t = 0; t < 8; ++t) b2c[t] = b2[kcol0 + 16 * t];

#pragma unroll
    for (int t = 0; t < 8; ++t)
#pragma unroll
        for (int i = 0; i < 8; ++i)
            y[t][i] = y0[(grow0 + crow0 + i) * DDIM + kcol0 + 16 * t];

    // Fused f(y): reads yh (LDS f16), accumulates raw tanh(y@W1+b1)@W2 into acc.
    auto feval = [&]() {
#pragma unroll
        for (int t = 0; t < 8; ++t)
#pragma unroll
            for (int i = 0; i < 8; ++i) acc[t][i] = 0.0f;

        for (int hc = 0; hc < HDIM / HC; ++hc) {
            // ---- GEMM1: a = y @ W1[:, chunk]  (one 16x16 tile per wave) ----
            const int ncol = hc * HC + ct * 16 + l16;         // B column (lane)
            const _Float16* bp = w1t + (long)ncol * DDIM + lhalf * 8;
            const _Float16* ap = yh + rowA * DDIM + lhalf * 8;

            if (hc + 1 < HDIM / HC) {   // pull next chunk's weights toward WGP
                __builtin_prefetch(w1t + (long)(ncol + HC) * DDIM, 0, 1);
                __builtin_prefetch(w2t + (long)kcol0 * HDIM + (hc + 1) * HC, 0, 1);
            }

            v8f a;
#pragma unroll
            for (int i = 0; i < 8; ++i) a[i] = 0.0f;
#pragma unroll
            for (int kk = 0; kk < DDIM / 32; ++kk) {
                v16h af = ld_frag(ap + 32 * kk,      ap + 32 * kk + 16);
                v16h bf = ld_frag(bp + 32 * kk,      bp + 32 * kk + 16);
                a = __builtin_amdgcn_wmma_f32_16x16x32_f16(
                        false, af, false, bf, (short)0, a, false, false);
            }

            // ---- bias + tanh -> f16 chunk in LDS (row-major [32 x 64]) ----
            const float b1v = b1[ncol];
#pragma unroll
            for (int i = 0; i < 8; ++i) {
                float hv = tanhf(a[i] + b1v);
                hb[(crow0 + i) * HC + ct * 16 + l16] = (_Float16)hv;
            }
            __syncthreads();

            // ---- GEMM2 partial: acc += h_chunk @ W2[chunk, :] ----
            const _Float16* hap = hb + (rt * 16 + l16) * HC + lhalf * 8;
#pragma unroll
            for (int t = 0; t < 8; ++t) {
                const _Float16* b2p =
                    w2t + (long)(kcol0 + 16 * t) * HDIM + hc * HC + lhalf * 8;
#pragma unroll
                for (int k2 = 0; k2 < HC / 32; ++k2) {
                    v16h af = ld_frag(hap + 32 * k2, hap + 32 * k2 + 16);
                    v16h bf = ld_frag(b2p + 32 * k2, b2p + 32 * k2 + 16);
                    acc[t] = __builtin_amdgcn_wmma_f32_16x16x32_f16(
                                 false, af, false, bf, (short)0, acc[t], false, false);
                }
            }
            __syncthreads();   // hb reusable next chunk
        }
    };

    for (int step = 0; step < NSTEPS; ++step) {
        // stage y -> yh (f16)
#pragma unroll
        for (int t = 0; t < 8; ++t)
#pragma unroll
            for (int i = 0; i < 8; ++i)
                yh[(crow0 + i) * DDIM + kcol0 + 16 * t] = (_Float16)y[t][i];
        __syncthreads();

        feval();                                  // acc = raw f-parts for k1

        // k1 = acc + b2; stage y2 = y + dt*k1; fold y += dt/2*k1 (k1 freed)
#pragma unroll
        for (int t = 0; t < 8; ++t)
#pragma unroll
            for (int i = 0; i < 8; ++i) {
                float k1 = acc[t][i] + b2c[t];
                yh[(crow0 + i) * DDIM + kcol0 + 16 * t] =
                    (_Float16)(y[t][i] + DT * k1);
                y[t][i] += 0.5f * DT * k1;
            }
        __syncthreads();

        feval();                                  // acc = raw f-parts for k2

#pragma unroll
        for (int t = 0; t < 8; ++t)
#pragma unroll
            for (int i = 0; i < 8; ++i)
                y[t][i] += 0.5f * DT * (acc[t][i] + b2c[t]);
    }

#pragma unroll
    for (int t = 0; t < 8; ++t)
#pragma unroll
        for (int i = 0; i < 8; ++i)
            yout[(grow0 + crow0 + i) * DDIM + kcol0 + 16 * t] = y[t][i];
}

// ---------------------------------------------------------------------------
extern "C" void kernel_launch(void* const* d_in, const int* in_sizes, int n_in,
                              void* d_out, int out_size, void* d_ws, size_t ws_size,
                              hipStream_t stream) {
    const float* y0 = (const float*)d_in[0];
    const float* W1 = (const float*)d_in[1];
    const float* b1 = (const float*)d_in[2];
    const float* W2 = (const float*)d_in[3];
    const float* b2 = (const float*)d_in[4];
    float* out = (float*)d_out;

    // scratch: W1T (2 MiB f16) + W2T (2 MiB f16)
    _Float16* w1t = (_Float16*)d_ws;
    _Float16* w2t = (_Float16*)((char*)d_ws + (size_t)HDIM * DDIM * sizeof(_Float16));

    const int nW = HDIM * DDIM;
    cvt_w1t<<<(nW + 255) / 256, 256, 0, stream>>>(W1, w1t);
    cvt_w2t<<<(nW + 255) / 256, 256, 0, stream>>>(W2, w2t);

    node_heun<<<BATCH / MT, 256, 0, stream>>>(y0, b1, b2, w1t, w2t, out);
}
